// TUPABlock_52080773431847
// MI455X (gfx1250) — matrix operand order repaired and checked by reading, not compiled, not run on hardware
//
#include <hip/hip_runtime.h>
#include <hip/hip_bf16.h>
#include <math.h>

// ---------------------------------------------------------------------------
// TUPA block for MI455X (gfx1250). All large contractions run through
// v_wmma_f32_16x16x32_f16 (wave32, 16x16 tiles, fp32 accumulate).
// The topo conv stack (cd_*) is mathematically dead: its bias is constant
// along the softmax axis, so it is skipped exactly.
// ---------------------------------------------------------------------------

typedef _Float16 half_t;
typedef __attribute__((ext_vector_type(8)))  _Float16 v8h;
typedef __attribute__((ext_vector_type(16))) _Float16 v16h;
typedef __attribute__((ext_vector_type(8)))  float    v8f;

#define NC    768          // N == C == 768
#define PD    64           // low-rank dim
#define KCONV (27 * 768)   // implicit-GEMM K for the 3x3x3 conv
#define LN_EPS 1e-5f
#define BN_EPS 1e-5f

__device__ __forceinline__ v16h cat8(v8h lo, v8h hi) {
  return __builtin_shufflevector(lo, hi, 0,1,2,3,4,5,6,7,8,9,10,11,12,13,14,15);
}

// ---------------------------------------------------------------------------
// Generic WMMA GEMM:  D[M,N] = A[M,K] * BT[N,K]^T  (+bias +relu +residual)
// A row-major f16, BT = B transposed, row-major f16 (16 contiguous K-halves
// per B-fragment lane load). Block = 128 threads = 4 waves; each wave owns a
// 16x64 strip (4 accumulators reuse one A fragment). M,N mult of 64; K of 32.
// Epilogue can emit f32, f16, and f16-transposed copies.
// ---------------------------------------------------------------------------
__global__ __launch_bounds__(128) void gemm_wmma_f16(
    const half_t* __restrict__ A, const half_t* __restrict__ BT,
    float* __restrict__ Cf, half_t* __restrict__ Ch, half_t* __restrict__ ChT,
    const float* __restrict__ bias, int biasMode /*0 none,1 per-M,2 per-N*/,
    int doRelu, const float* __restrict__ residual, int M, int N, int K)
{
  const int lane  = threadIdx.x & 31;
  const int wave  = threadIdx.x >> 5;
  const int lm    = lane & 15;
  const int lh    = lane >> 4;
  const int mBase = blockIdx.y * 64 + wave * 16;
  const int nBase = blockIdx.x * 64;

  const half_t* aRow  = A  + (size_t)(mBase + lm) * K;
  const half_t* bRow0 = BT + (size_t)(nBase + lm) * K + 16 * lh;

  v8f acc[4] = {};
  for (int k0 = 0; k0 < K; k0 += 32) {
    v8h alo = *(const v8h*)(aRow + k0 + 16 * lh);
    v8h ahi = *(const v8h*)(aRow + k0 + 8 + 16 * lh);
    v16h a  = cat8(alo, ahi);
#pragma unroll
    for (int j = 0; j < 4; ++j) {
      v16h b = *(const v16h*)(bRow0 + (size_t)j * 16 * K + k0);
      acc[j] = __builtin_amdgcn_wmma_f32_16x16x32_f16(
          false, a, false, b, (short)0, acc[j], false, false);
    }
  }
#pragma unroll
  for (int j = 0; j < 4; ++j) {
    const int n = nBase + j * 16 + lm;
#pragma unroll
    for (int r = 0; r < 8; ++r) {
      const int m = mBase + r + 8 * lh;   // C/D layout: VGPR r -> M=r (+8 hi half)
      float v = acc[j][r];
      if (biasMode == 1) v += bias[m];
      else if (biasMode == 2) v += bias[n];
      if (residual) v += residual[(size_t)m * N + n];
      if (doRelu) v = fmaxf(v, 0.0f);
      if (Cf)  Cf [(size_t)m * N + n] = v;
      if (Ch)  Ch [(size_t)m * N + n] = (half_t)v;
      if (ChT) ChT[(size_t)n * M + m] = (half_t)v;
    }
  }
}

// ---------------------------------------------------------------------------
// 3x3x3 conv as implicit WMMA GEMM over (co=768) x (sp=768), K = 27*768.
// Wr[co][t*768+ci] = rf_w1[co][ci][t];  XfT[sp][ci] (row 768 = zeros for pad).
// Spatial dims (12,8,8), pad 1, cross-correlation (no kernel flip).
// ---------------------------------------------------------------------------
__global__ __launch_bounds__(128) void conv3_wmma_f16(
    const half_t* __restrict__ Wr, const half_t* __restrict__ XfT,
    float* __restrict__ Out, const float* __restrict__ bias)
{
  const int lane  = threadIdx.x & 31;
  const int wave  = threadIdx.x >> 5;
  const int lm    = lane & 15;
  const int lh    = lane >> 4;
  const int mBase = blockIdx.y * 64 + wave * 16;
  const int nBase = blockIdx.x * 64;

  const half_t* aRow = Wr + (size_t)(mBase + lm) * KCONV;
  v8f acc[4] = {};

  for (int t = 0; t < 27; ++t) {
    const int t0 = t / 9, t1 = (t / 3) % 3, t2 = t % 3;
    const half_t* bRow[4];
#pragma unroll
    for (int j = 0; j < 4; ++j) {
      const int sp = nBase + j * 16 + lm;
      const int s0 = sp >> 6, s1 = (sp >> 3) & 7, s2 = sp & 7;
      const int a0 = s0 + t0 - 1, a1 = s1 + t1 - 1, a2 = s2 + t2 - 1;
      const bool ok = (unsigned)a0 < 12u && (unsigned)a1 < 8u && (unsigned)a2 < 8u;
      const int nsp = ok ? ((a0 << 6) | (a1 << 3) | a2) : 768;  // 768 = zero row
      bRow[j] = XfT + (size_t)nsp * NC + 16 * lh;
    }
    const half_t* aT = aRow + t * NC;
    for (int k0 = 0; k0 < NC; k0 += 32) {
      v8h alo = *(const v8h*)(aT + k0 + 16 * lh);
      v8h ahi = *(const v8h*)(aT + k0 + 8 + 16 * lh);
      v16h a  = cat8(alo, ahi);
#pragma unroll
      for (int j = 0; j < 4; ++j) {
        v16h b = *(const v16h*)(bRow[j] + k0);
        acc[j] = __builtin_amdgcn_wmma_f32_16x16x32_f16(
            false, a, false, b, (short)0, acc[j], false, false);
      }
    }
  }
#pragma unroll
  for (int j = 0; j < 4; ++j) {
    const int n = nBase + j * 16 + lm;
#pragma unroll
    for (int r = 0; r < 8; ++r) {
      const int m = mBase + r + 8 * lh;
      Out[(size_t)m * NC + n] = acc[j][r] + bias[m];
    }
  }
}

// ---------------------------------------------------------------------------
// LayerNorm over channels: x is (C,spatial) flat; x_flat[n][c] = x[c*768+n].
// Emits fp32, f16 and f16-transposed normalized activations.
// ---------------------------------------------------------------------------
__global__ __launch_bounds__(256) void layernorm_kernel(
    const float* __restrict__ x, const float* __restrict__ gamma,
    const float* __restrict__ beta, float* __restrict__ Xn,
    half_t* __restrict__ Xn_h, half_t* __restrict__ XnT_h)
{
  __shared__ float s1[256], s2[256];
  const int n = blockIdx.x, tid = threadIdx.x;
  float v[3], a = 0.f, b = 0.f;
#pragma unroll
  for (int i = 0; i < 3; ++i) {
    const int c = tid + i * 256;
    const float t = x[(size_t)c * NC + n];
    v[i] = t; a += t; b += t * t;
  }
  s1[tid] = a; s2[tid] = b; __syncthreads();
  for (int off = 128; off > 0; off >>= 1) {
    if (tid < off) { s1[tid] += s1[tid + off]; s2[tid] += s2[tid + off]; }
    __syncthreads();
  }
  const float mean = s1[0] * (1.0f / NC);
  const float var  = s2[0] * (1.0f / NC) - mean * mean;   // biased (torch LN)
  const float rstd = rsqrtf(var + LN_EPS);
#pragma unroll
  for (int i = 0; i < 3; ++i) {
    const int c = tid + i * 256;
    const float y = (v[i] - mean) * rstd * gamma[c] + beta[c];
    Xn  [(size_t)n * NC + c] = y;
    Xn_h[(size_t)n * NC + c] = (half_t)y;
    XnT_h[(size_t)c * NC + n] = (half_t)y;
  }
}

// Row softmax with pre-scale; optional transposed f16 store (for BT operands).
__global__ __launch_bounds__(256) void softmax_rows(
    const float* __restrict__ in, half_t* __restrict__ out,
    int rows, int L, float scale, int transposed)
{
  __shared__ float red[256];
  const int row = blockIdx.x, tid = threadIdx.x;
  const float* rp = in + (size_t)row * L;
  float m = -3.4e38f;
  for (int j = tid; j < L; j += 256) m = fmaxf(m, rp[j] * scale);
  red[tid] = m; __syncthreads();
  for (int off = 128; off > 0; off >>= 1) {
    if (tid < off) red[tid] = fmaxf(red[tid], red[tid + off]);
    __syncthreads();
  }
  m = red[0]; __syncthreads();
  float s = 0.f;
  for (int j = tid; j < L; j += 256) s += expf(rp[j] * scale - m);
  red[tid] = s; __syncthreads();
  for (int off = 128; off > 0; off >>= 1) {
    if (tid < off) red[tid] += red[tid + off];
    __syncthreads();
  }
  const float inv = 1.0f / red[0];
  for (int j = tid; j < L; j += 256) {
    const float e = expf(rp[j] * scale - m) * inv;
    if (transposed) out[(size_t)j * rows + row] = (half_t)e;
    else            out[(size_t)row * L + j]    = (half_t)e;
  }
}

// --- small utility kernels --------------------------------------------------
__global__ void transpose_f32_f16(const float* __restrict__ in,
                                  half_t* __restrict__ out, int R, int S) {
  const int idx = blockIdx.x * 256 + threadIdx.x;
  if (idx >= R * S) return;
  const int r = idx / S, s = idx - r * S;
  out[(size_t)s * R + r] = (half_t)in[idx];
}
__global__ void convert_f32_f16(const float* __restrict__ in,
                                half_t* __restrict__ out, int n) {
  const int idx = blockIdx.x * 256 + threadIdx.x;
  if (idx < n) out[idx] = (half_t)in[idx];
}
__global__ void zero_f16(half_t* __restrict__ p, int n) {
  const int idx = blockIdx.x * 256 + threadIdx.x;
  if (idx < n) p[idx] = (half_t)0.0f;
}
// rf_w1 (O,I,3,3,3) -> Wr[co][t*768+ci]
__global__ void reorder_w1(const float* __restrict__ in, half_t* __restrict__ out) {
  const int idx = blockIdx.x * 256 + threadIdx.x;
  if (idx >= NC * NC * 27) return;
  const int co = idx / (NC * 27);
  const int rem = idx - co * NC * 27;
  const int ci = rem / 27, t = rem - ci * 27;
  out[(size_t)co * KCONV + t * NC + ci] = (half_t)in[idx];
}
// BN(eval)+ReLU, transposed f16 store: rT[sp][co]
__global__ void bn_relu_transpose(const float* __restrict__ r,
    const float* __restrict__ g, const float* __restrict__ b,
    const float* __restrict__ mean, const float* __restrict__ var,
    half_t* __restrict__ rT) {
  const int idx = blockIdx.x * 256 + threadIdx.x;
  if (idx >= NC * NC) return;
  const int co = idx / NC, sp = idx - co * NC;
  float v = (r[idx] - mean[co]) * rsqrtf(var[co] + BN_EPS) * g[co] + b[co];
  rT[(size_t)sp * NC + co] = (half_t)fmaxf(v, 0.0f);
}
// uncertainty head: u1(relu'd, N x 384) @ w_u2(384,2) -> softplus -> gates
__global__ void uncertainty_kernel(const float* __restrict__ u1,
    const float* __restrict__ w_u2, const float* __restrict__ b_u2,
    float* __restrict__ g_sc, float* __restrict__ g_topo,
    float* __restrict__ out_unc) {
  const int n = blockIdx.x * 64 + threadIdx.x;
  if (n >= NC) return;
  const float* u = u1 + (size_t)n * 384;
  float z0 = b_u2[0], z1 = b_u2[1];
  for (int j = 0; j < 384; ++j) {
    const float t = u[j];
    z0 += t * w_u2[2 * j];
    z1 += t * w_u2[2 * j + 1];
  }
  const float u0 = (z0 > 20.f) ? z0 : log1pf(expf(z0));
  const float u1v = (z1 > 20.f) ? z1 : log1pf(expf(z1));
  const float sig = u0 + u1v;
  const float gt = 1.0f / (1.0f + expf(-sig));   // sigmoid(sigma)
  g_topo[n] = gt; g_sc[n] = 1.0f - gt;           // sigmoid(-sigma)
  out_unc[n] = u0; out_unc[NC + n] = u1v;
}
// gated fusion; transposed store (XfT[sp=c][ci=n]) = conv input layout
__global__ void fuse_kernel(const float* __restrict__ xs,
    const float* __restrict__ xc, const float* __restrict__ xt,
    const float* __restrict__ g_sc, const float* __restrict__ g_topo,
    half_t* __restrict__ XfT) {
  const int idx = blockIdx.x * 256 + threadIdx.x;
  if (idx >= NC * NC) return;
  const int n = idx / NC, c = idx - n * NC;
  const float v = g_sc[n] * (xs[idx] + xc[idx]) + g_topo[n] * xt[idx];
  XfT[(size_t)c * NC + n] = (half_t)v;
}

// ---------------------------------------------------------------------------
extern "C" void kernel_launch(void* const* d_in, const int* in_sizes, int n_in,
                              void* d_out, int out_size, void* d_ws, size_t ws_size,
                              hipStream_t stream) {
  (void)in_sizes; (void)n_in; (void)out_size; (void)ws_size;
  const float* x     = (const float*)d_in[0];
  const float* ln_g  = (const float*)d_in[1];
  const float* ln_b  = (const float*)d_in[2];
  const float* w_qk  = (const float*)d_in[3];
  const float* b_qk  = (const float*)d_in[4];
  const float* w_sp  = (const float*)d_in[5];
  const float* b_sp  = (const float*)d_in[6];
  const float* w_vs  = (const float*)d_in[7];
  const float* b_vs  = (const float*)d_in[8];
  const float* w_vc  = (const float*)d_in[9];
  const float* b_vc  = (const float*)d_in[10];
  const float* w_vt  = (const float*)d_in[11];
  const float* b_vt  = (const float*)d_in[12];
  // d_in[13..18] = cd_* conv stack: exactly dead (row-constant softmax bias)
  const float* w_u1  = (const float*)d_in[19];
  const float* b_u1  = (const float*)d_in[20];
  const float* w_u2  = (const float*)d_in[21];
  const float* b_u2  = (const float*)d_in[22];
  const float* rf_w1 = (const float*)d_in[23];
  const float* rf_b1 = (const float*)d_in[24];
  const float* bn_g  = (const float*)d_in[25];
  const float* bn_b  = (const float*)d_in[26];
  const float* bn_m  = (const float*)d_in[27];
  const float* bn_v  = (const float*)d_in[28];
  const float* rf_w2 = (const float*)d_in[29];
  const float* rf_b2 = (const float*)d_in[30];
  float* out = (float*)d_out;

  char* ws = (char*)d_ws;
  size_t off = 0;
  auto alloc = [&](size_t bytes) -> void* {
    size_t a = (off + 255) & ~(size_t)255;
    off = a + bytes;
    return ws + a;
  };
  const size_t E = (size_t)NC * NC;           // 589824
  float* Xn      = (float*)alloc(E * 4);
  float* logitsS = (float*)alloc((size_t)NC * PD * 4);
  float* logitsC = (float*)alloc(E * 4);
  float* logitsT = (float*)alloc(E * 4);
  float* xsb     = (float*)alloc(E * 4);
  float* xcb     = (float*)alloc(E * 4);
  float* xtb     = (float*)alloc(E * 4);
  float* u1buf   = (float*)alloc((size_t)NC * 384 * 4);
  float* r_raw   = (float*)alloc(E * 4);
  float* g_sc    = (float*)alloc(NC * 4);
  float* g_topo  = (float*)alloc(NC * 4);
  half_t* Xn_h   = (half_t*)alloc(E * 2);
  half_t* XnT_h  = (half_t*)alloc(E * 2);
  half_t* wqkT   = (half_t*)alloc(E * 2);
  half_t* wvsT   = (half_t*)alloc(E * 2);
  half_t* wvcT   = (half_t*)alloc(E * 2);
  half_t* wvtT   = (half_t*)alloc(E * 2);
  half_t* wspT   = (half_t*)alloc((size_t)NC * PD * 2);
  half_t* wu1T   = (half_t*)alloc((size_t)NC * 384 * 2);
  half_t* qk_h   = (half_t*)alloc(E * 2);
  half_t* qkT_h  = (half_t*)alloc(E * 2);
  half_t* vsT_h  = (half_t*)alloc(E * 2);
  half_t* vc_h   = (half_t*)alloc(E * 2);
  half_t* vtT_h  = (half_t*)alloc(E * 2);
  half_t* kprj_h = (half_t*)alloc((size_t)PD * NC * 2);
  half_t* vspT_h = (half_t*)alloc((size_t)NC * PD * 2);
  half_t* atS_h  = (half_t*)alloc((size_t)NC * PD * 2);
  half_t* atCT_h = (half_t*)alloc(E * 2);
  half_t* atT_h  = (half_t*)alloc(E * 2);
  half_t* XfT_h  = (half_t*)alloc((size_t)769 * NC * 2);  // +1 zero row for pad
  half_t* w1r_h  = (half_t*)alloc((size_t)NC * KCONV * 2);
  half_t* rT_h   = (half_t*)alloc(E * 2);
  half_t* w2_h   = (half_t*)alloc(E * 2);

  const float scl = 1.0f / sqrtf((float)NC);
  auto gemm = [&](const half_t* A, const half_t* BT, float* Cf, half_t* Ch,
                  half_t* ChT, const float* bias, int bm, int relu,
                  const float* resid, int M, int N, int K) {
    dim3 g(N / 64, M / 64);
    gemm_wmma_f16<<<g, 128, 0, stream>>>(A, BT, Cf, Ch, ChT, bias, bm, relu,
                                         resid, M, N, K);
  };
  auto blocks = [](size_t n) { return dim3((unsigned)((n + 255) / 256)); };

  // --- weight prep (f16, transposed for BT usage) ---
  transpose_f32_f16<<<blocks(E), 256, 0, stream>>>(w_qk, wqkT, NC, NC);
  transpose_f32_f16<<<blocks(E), 256, 0, stream>>>(w_vs, wvsT, NC, NC);
  transpose_f32_f16<<<blocks(E), 256, 0, stream>>>(w_vc, wvcT, NC, NC);
  transpose_f32_f16<<<blocks(E), 256, 0, stream>>>(w_vt, wvtT, NC, NC);
  transpose_f32_f16<<<blocks((size_t)NC * PD), 256, 0, stream>>>(w_sp, wspT, NC, PD);
  transpose_f32_f16<<<blocks((size_t)NC * 384), 256, 0, stream>>>(w_u1, wu1T, NC, 384);
  convert_f32_f16<<<blocks(E), 256, 0, stream>>>(rf_w2, w2_h, (int)E);
  reorder_w1<<<blocks((size_t)NC * KCONV), 256, 0, stream>>>(rf_w1, w1r_h);
  zero_f16<<<blocks(NC), 256, 0, stream>>>(XfT_h + (size_t)768 * NC, NC);

  // --- LayerNorm ---
  layernorm_kernel<<<NC, 256, 0, stream>>>(x, ln_g, ln_b, Xn, Xn_h, XnT_h);

  // --- projections (all A = Xn_h) ---
  gemm(Xn_h, wqkT, nullptr, qk_h, qkT_h, b_qk, 2, 0, nullptr, NC, NC, NC); // qk
  gemm(Xn_h, wvsT, nullptr, nullptr, vsT_h, b_vs, 2, 0, nullptr, NC, NC, NC); // v_spatial^T
  gemm(Xn_h, wvcT, nullptr, vc_h, nullptr, b_vc, 2, 0, nullptr, NC, NC, NC); // v_channel
  gemm(Xn_h, wvtT, nullptr, nullptr, vtT_h, b_vt, 2, 0, nullptr, NC, NC, NC); // v_topo^T
  gemm(Xn_h, wu1T, u1buf, nullptr, nullptr, b_u1, 2, 1, nullptr, NC, 384, NC); // relu(u1)

  // --- low-rank spatial attention ---
  gemm(wspT, qkT_h, nullptr, kprj_h, nullptr, b_sp, 1, 0, nullptr, PD, NC, NC); // k_proj (P,C)
  gemm(vsT_h, wspT, nullptr, vspT_h, nullptr, b_sp, 2, 0, nullptr, NC, PD, NC); // v_sp_proj^T (C,P)
  gemm(qk_h, kprj_h, logitsS, nullptr, nullptr, nullptr, 0, 0, nullptr, NC, PD, NC);
  softmax_rows<<<NC, 256, 0, stream>>>(logitsS, atS_h, NC, PD, scl, 0);
  gemm(atS_h, vspT_h, xsb, nullptr, nullptr, nullptr, 0, 0, nullptr, NC, NC, PD); // x_spatial

  // --- channel attention ---
  gemm(qkT_h, qkT_h, logitsC, nullptr, nullptr, nullptr, 0, 0, nullptr, NC, NC, NC);
  softmax_rows<<<NC, 256, 0, stream>>>(logitsC, atCT_h, NC, NC, scl, 1);
  gemm(vc_h, atCT_h, xcb, nullptr, nullptr, nullptr, 0, 0, nullptr, NC, NC, NC); // x_channel

  // --- topology attention (bias row-constant => exact softmax no-op) ---
  gemm(qk_h, qk_h, logitsT, nullptr, nullptr, nullptr, 0, 0, nullptr, NC, NC, NC);
  softmax_rows<<<NC, 256, 0, stream>>>(logitsT, atT_h, NC, NC, scl, 0);
  gemm(atT_h, vtT_h, xtb, nullptr, nullptr, nullptr, 0, 0, nullptr, NC, NC, NC); // x_topo

  // --- uncertainty gates + fusion (writes conv input transposed) ---
  uncertainty_kernel<<<12, 64, 0, stream>>>(u1buf, w_u2, b_u2, g_sc, g_topo,
                                            out + E);
  fuse_kernel<<<blocks(E), 256, 0, stream>>>(xsb, xcb, xtb, g_sc, g_topo, XfT_h);

  // --- refine: 3x3x3 conv (implicit WMMA GEMM) + BN + ReLU + 1x1 conv + res ---
  {
    dim3 g(NC / 64, NC / 64);
    conv3_wmma_f16<<<g, 128, 0, stream>>>(w1r_h, XfT_h, r_raw, rf_b1);
  }
  bn_relu_transpose<<<blocks(E), 256, 0, stream>>>(r_raw, bn_g, bn_b, bn_m, bn_v, rT_h);
  gemm(w2_h, rT_h, out, nullptr, nullptr, rf_b2, 1, 0, x, NC, NC, NC); // -> d_out
}
